// MedTrans_21268678050491
// MI455X (gfx1250) — matrix operand order repaired and checked by reading, not compiled, not run on hardware
//
#include <hip/hip_runtime.h>
#include <hip/hip_bf16.h>

// ---------------------------------------------------------------------------
// MedTrans forward for MI455X (gfx1250): bf16 WMMA GEMMs, fp32 accumulate.
// Only batch 0 of the attention path is computed (reference uses ao[0] only).
// Weights are cast fp32->bf16 *transposed* so WMMA A/B fragments are
// contiguous 16B loads per lane (CDNA5 16-bit A 16x32 / B 32x16 layouts).
// GEMM: block tile 128(M) x NF*16(N) (NF=8 normally, NF=4 when N==64),
// 4 wave32 waves, 32 x NF*16 per wave; B tile staged in LDS (double-buffered,
// one barrier/K-step); B fragments ping-pong between two register sets so
// ds_load of fragment f+1 overlaps the WMMAs on fragment f.
// ---------------------------------------------------------------------------

typedef __attribute__((ext_vector_type(16))) __bf16 v16bf;
typedef __attribute__((ext_vector_type(8)))  __bf16 v8bf;
typedef __attribute__((ext_vector_type(8)))  float  v8f;

using bstore = __hip_bfloat16;

union FragAB { v16bf v; v8bf h[2]; };

__device__ __forceinline__ bstore f2b(float f) { return __float2bfloat16(f); }

namespace {
constexpr int Bn = 16, Sn = 512, INn = 512, OUTn = 256;
constexpr int Dn = 1024, Hn = 16, Ln = 6, DFFn = 2048, HDn = 64, HIDn = 256;
constexpr int BSn = Bn * Sn;                 // 8192 tokens
constexpr float SCALEf = 0.125f;             // HD^-0.5 = 64^-0.5
constexpr int BSTRIDE = 40;                  // padded LDS row (bf16 elems) -> 80B
}

// ---------------------------------------------------------------------------
// fp32 -> bf16 straight cast
__global__ void __launch_bounds__(256)
k_cast_bf16(const float* __restrict__ in, bstore* __restrict__ out, long long n) {
  long long i = (long long)blockIdx.x * blockDim.x + threadIdx.x;
  if (i < n) out[i] = f2b(in[i]);
}

// fp32 (K x N row-major) -> bf16 transposed (N x K row-major)
__global__ void __launch_bounds__(256)
k_cast_t_bf16(const float* __restrict__ w, bstore* __restrict__ wt, int K, int N) {
  long long i = (long long)blockIdx.x * blockDim.x + threadIdx.x;
  long long total = (long long)K * N;
  if (i >= total) return;
  int n = (int)(i / K);
  int k = (int)(i - (long long)n * K);
  wt[i] = f2b(w[(long long)k * N + n]);
}

// key-padding mask flag for batch 0: (sum_s xm[0,s] == 0)
__global__ void __launch_bounds__(256)
k_mask(const float* __restrict__ xm, int S, int* __restrict__ flag) {
  __shared__ float red[256];
  float s = 0.f;
  for (int i = threadIdx.x; i < S; i += 256) s += xm[i];
  red[threadIdx.x] = s; __syncthreads();
  for (int o = 128; o > 0; o >>= 1) {
    if (threadIdx.x < o) red[threadIdx.x] += red[threadIdx.x + o];
    __syncthreads();
  }
  if (threadIdx.x == 0) *flag = (red[0] == 0.0f) ? 1 : 0;
}

// clamp-to-[-10,10] softmax over `cols` (256 threads/row, cols in {512,1024}).
// With the clamp, exp() is bounded -> no max-subtraction pass needed.
__global__ void __launch_bounds__(256)
k_softmax_row(const float* __restrict__ in, float* __restrict__ outF,
              bstore* __restrict__ outB, int cols, const int* __restrict__ maskflag) {
  __shared__ float red[256];
  const long long row = blockIdx.x;
  const float* x = in + row * cols;
  const int tid = threadIdx.x;
  const int per = cols >> 8;          // elements per thread (<=4)
  const int msk = maskflag ? *maskflag : 0;
  float ev[4];
  float lsum = 0.f;
  for (int j = 0; j < per; ++j) {
    int c = tid + (j << 8);
    float v = x[c];
    v = fminf(10.f, fmaxf(-10.f, v)); // -inf (masked) clips to -10
    if (msk) v = -10.f;
    float e = __expf(v);
    ev[j] = e; lsum += e;
  }
  red[tid] = lsum; __syncthreads();
  for (int o = 128; o > 0; o >>= 1) {
    if (tid < o) red[tid] += red[tid + o];
    __syncthreads();
  }
  float inv = 1.f / red[0];
  for (int j = 0; j < per; ++j) {
    int c = tid + (j << 8);
    float r = ev[j] * inv;
    if (outF) outF[row * cols + c] = r;
    if (outB) outB[row * cols + c] = f2b(r);
  }
}

// fused residual add + LayerNorm over D; writes fp32 h and bf16 h copy.
// addMask: S-1 broadcasts a (S,D) addend over batch; 0xFFFFFFFF = per-row.
__global__ void __launch_bounds__(256)
k_ln(const float* __restrict__ hin, const float* __restrict__ addv, unsigned addMask,
     const float* __restrict__ g, const float* __restrict__ b,
     float* __restrict__ hout, bstore* __restrict__ hb, int Dd) {
  __shared__ float red[256];
  const long long row = blockIdx.x;
  const float* x0 = hin + row * Dd;
  const float* a0 = addv + (long long)((unsigned)row & addMask) * Dd;
  const int tid = threadIdx.x;
  const int per = Dd >> 8;            // = 4 for D=1024
  float xv[4];
  float s = 0.f;
  for (int j = 0; j < per; ++j) { int c = tid + (j << 8); xv[j] = x0[c] + a0[c]; s += xv[j]; }
  red[tid] = s; __syncthreads();
  for (int o = 128; o > 0; o >>= 1) { if (tid < o) red[tid] += red[tid + o]; __syncthreads(); }
  float mean = red[0] / Dd; __syncthreads();
  float s2 = 0.f;
  for (int j = 0; j < per; ++j) { float d = xv[j] - mean; s2 += d * d; }
  red[tid] = s2; __syncthreads();
  for (int o = 128; o > 0; o >>= 1) { if (tid < o) red[tid] += red[tid + o]; __syncthreads(); }
  float inv = rsqrtf(red[0] / Dd + 1e-5f);
  for (int j = 0; j < per; ++j) {
    int c = tid + (j << 8);
    float y = (xv[j] - mean) * inv * g[c] + b[c];
    hout[row * Dd + c] = y;
    hb[row * Dd + c]  = f2b(y);
  }
}

// ---------------------------------------------------------------------------
// Generic bf16 WMMA GEMM: C = act( scale*(A @ Bt^T + bias [+ add]) ).
//   A : M x K row-major bf16 (stride lda), per-z offset sA elements
//   Bt: N x K row-major bf16 (B pre-transposed; stride ldb), per-z offset sB
// Block = 128 threads = 4 waves; tile 128(M) x 16*NF(N); wave w owns rows
// [mbase+32w, mbase+32w+32) as two 16-row WMMA sub-tiles sharing B fragments.
// B tile (16*NF x 32 bf16) staged in LDS, double-buffered with a single
// __syncthreads per K-step; next tile's global loads issued before compute;
// B fragments ping-pong between two register sets (no copies) so the
// ds_load_b128 pair for fragment f+1 overlaps the WMMA pair on fragment f.
// Requires: M % 128 == 0, N % (16*NF) == 0, K % 32 == 0.
// Fragment layouts (CDNA5 ISA 7.12.2, wave32):
//   A 16x32 : lanes 0-15 row m, K[kb..kb+7] + K[kb+16..kb+23];
//             lanes 16-31 row m, K[kb+8..kb+15] + K[kb+24..kb+31]
//   B 32x16 : same K split as A (lanes 0-15 K[kb..kb+7]+K[kb+16..kb+23]; 16-31 K[kb+8..kb+15]+K[kb+24..kb+31])
template <int NF>
__global__ void __launch_bounds__(128)
k_gemm_bf16(const bstore* __restrict__ A, int lda, long long sA,
            const bstore* __restrict__ Bt, int ldb, long long sB,
            const float* __restrict__ bias,
            const float* __restrict__ add, unsigned addMask, int addLd,
            float* __restrict__ Cf, int ldc, long long sCf,
            bstore* __restrict__ Cb, int ldcb, long long sCb,
            bstore* __restrict__ CbT, int ldcbt, long long sCbT,
            int M, int N, int K, float scale, int relu) {
  constexpr int TN  = NF * 16;   // block N-tile
  constexpr int CPT = TN / 32;   // v8bf staging chunks per thread
  __shared__ __align__(16) bstore Bs[2][TN * BSTRIDE];

  const int tid  = threadIdx.x;
  const int wv   = tid >> 5;           // wave id 0..3
  const int lane = tid & 31;
  const int hf   = lane >> 4;          // half-wave select
  const int l16  = lane & 15;
  const int z    = blockIdx.z;
  const bstore* Az = A  + (long long)z * sA;
  const bstore* Bz = Bt + (long long)z * sB;
  const int mbase = blockIdx.x * 128 + wv * 32;
  const int nbase = blockIdx.y * TN;

  v8f acc[2][NF] = {};

  // A fragment row pointers (per-wave private rows; hf chunk offset pre-added)
  const bstore* arow0 = Az + (long long)(mbase + l16) * lda + (hf << 3);
  const bstore* arow1 = arow0 + (long long)16 * lda;

  // Cooperative B staging: chunk ci = tid + 128*c -> row ci>>2, k-chunk (ci&3)*8
  const bstore* bsrc[CPT];
  int bofs[CPT];
#pragma unroll
  for (int c = 0; c < CPT; ++c) {
    const int ci  = tid + (c << 7);
    const int row = ci >> 2;
    const int kc  = (ci & 3) << 3;
    bsrc[c] = Bz + (long long)(nbase + row) * ldb + kc;
    bofs[c] = row * BSTRIDE + kc;
  }

  { // prologue: stage K-tile 0 into buffer 0
#pragma unroll
    for (int c = 0; c < CPT; ++c) {
      v8bf t = *reinterpret_cast<const v8bf*>(bsrc[c]);
      *reinterpret_cast<v8bf*>(&Bs[0][bofs[c]]) = t;
    }
  }
  __syncthreads();

  int cur = 0;
  for (int kb = 0; kb < K; kb += 32) {
    const bool hasNext = (kb + 32) < K;   // uniform across block
    v8bf nx[CPT];
    if (hasNext) {                        // issue next-tile loads early
#pragma unroll
      for (int c = 0; c < CPT; ++c)
        nx[c] = *reinterpret_cast<const v8bf*>(bsrc[c] + kb + 32);
    }
    FragAB a0, a1;
    const bstore* ap0 = arow0 + kb;
    const bstore* ap1 = arow1 + kb;
    a0.h[0] = *reinterpret_cast<const v8bf*>(ap0);
    a0.h[1] = *reinterpret_cast<const v8bf*>(ap0 + 16);
    a1.h[0] = *reinterpret_cast<const v8bf*>(ap1);
    a1.h[1] = *reinterpret_cast<const v8bf*>(ap1 + 16);
    __builtin_prefetch((const void*)(ap0 + 32), 0, 3);  // next K-tile of A
    __builtin_prefetch((const void*)(ap1 + 32), 0, 3);

    // B fragments: two distinct register sets, ping-pong by f parity; the
    // loads for fragment f+1 are issued before the WMMAs on fragment f.
    FragAB bf[2];
    {
      const bstore* bp = &Bs[cur][l16 * BSTRIDE + (hf << 3)];
      bf[0].h[0] = *reinterpret_cast<const v8bf*>(bp);
      bf[0].h[1] = *reinterpret_cast<const v8bf*>(bp + 16);
    }
#pragma unroll
    for (int f = 0; f < NF; ++f) {
      if (f + 1 < NF) {
        const bstore* bp = &Bs[cur][(((f + 1) << 4) + l16) * BSTRIDE + (hf << 3)];
        bf[(f + 1) & 1].h[0] = *reinterpret_cast<const v8bf*>(bp);
        bf[(f + 1) & 1].h[1] = *reinterpret_cast<const v8bf*>(bp + 16);
      }
      acc[0][f] = __builtin_amdgcn_wmma_f32_16x16x32_bf16(
          false, a0.v, false, bf[f & 1].v, (short)0, acc[0][f], false, false);
      acc[1][f] = __builtin_amdgcn_wmma_f32_16x16x32_bf16(
          false, a1.v, false, bf[f & 1].v, (short)0, acc[1][f], false, false);
    }

    if (hasNext) {
#pragma unroll
      for (int c = 0; c < CPT; ++c)
        *reinterpret_cast<v8bf*>(&Bs[cur ^ 1][bofs[c]]) = nx[c];
      cur ^= 1;
    }
    __syncthreads();  // single barrier per K-step (buffers alternate)
  }

  // Epilogue. C/D 16x16 f32 layout: VGPR r, lanes 0-15 -> M=r, lanes 16-31 -> M=8+r.
#pragma unroll
  for (int su = 0; su < 2; ++su) {
#pragma unroll
    for (int f = 0; f < NF; ++f) {
      const int n = nbase + (f << 4) + l16;
      const float bv = bias ? bias[n] : 0.f;
#pragma unroll
      for (int r = 0; r < 8; ++r) {
        const int m = mbase + (su << 4) + (hf << 3) + r;
        float v = acc[su][f][r] + bv;
        if (add) v += add[(long long)((unsigned)m & addMask) * addLd + n];
        v *= scale;
        if (relu) v = fmaxf(v, 0.f);
        if (Cf)  Cf [(long long)z * sCf  + (long long)m * ldc   + n] = v;
        if (Cb)  Cb [(long long)z * sCb  + (long long)m * ldcb  + n] = f2b(v);
        if (CbT) CbT[(long long)z * sCbT + (long long)n * ldcbt + m] = f2b(v);
      }
    }
  }
}

// ---------------------------------------------------------------------------
// host-side helpers
static void launch_gemm(hipStream_t s,
                        const bstore* A, int lda, long long sA,
                        const bstore* Bt, int ldb, long long sB,
                        const float* bias, const float* add, unsigned addMask, int addLd,
                        float* Cf, int ldc, long long sCf,
                        bstore* Cb, int ldcb, long long sCb,
                        bstore* CbT, int ldcbt, long long sCbT,
                        int M, int N, int K, float scale, int relu, int Z) {
  dim3 block(128);
  if ((N & 127) == 0) {
    dim3 grid(M / 128, N / 128, Z);
    k_gemm_bf16<8><<<grid, block, 0, s>>>(A, lda, sA, Bt, ldb, sB, bias, add, addMask,
                                          addLd, Cf, ldc, sCf, Cb, ldcb, sCb,
                                          CbT, ldcbt, sCbT, M, N, K, scale, relu);
  } else {
    dim3 grid(M / 128, N / 64, Z);
    k_gemm_bf16<4><<<grid, block, 0, s>>>(A, lda, sA, Bt, ldb, sB, bias, add, addMask,
                                          addLd, Cf, ldc, sCf, Cb, ldcb, sCb,
                                          CbT, ldcbt, sCbT, M, N, K, scale, relu);
  }
}

static void launch_castT(hipStream_t s, const float* w, bstore* wt, int K, int N) {
  long long n = (long long)K * N;
  k_cast_t_bf16<<<(unsigned)((n + 255) / 256), 256, 0, s>>>(w, wt, K, N);
}

extern "C" void kernel_launch(void* const* d_in, const int* in_sizes, int n_in,
                              void* d_out, int out_size, void* d_ws, size_t ws_size,
                              hipStream_t stream) {
  (void)in_sizes; (void)n_in; (void)out_size; (void)ws_size;
  const float* x      = (const float*)d_in[0];
  const float* xm     = (const float*)d_in[1];
  /* d_in[2] vecSelector: assert-only, unused */
  const float* proj_w = (const float*)d_in[3];
  const float* proj_b = (const float*)d_in[4];
  const float* pos    = (const float*)d_in[5];
  const float* qw = (const float*)d_in[6],  *qb = (const float*)d_in[7];
  const float* kw = (const float*)d_in[8],  *kb = (const float*)d_in[9];
  const float* vw = (const float*)d_in[10], *vb = (const float*)d_in[11];
  const float* ow = (const float*)d_in[12], *ob = (const float*)d_in[13];
  const float* ln1_g = (const float*)d_in[14], *ln1_b = (const float*)d_in[15];
  const float* ln2_g = (const float*)d_in[16], *ln2_b = (const float*)d_in[17];
  const float* w1 = (const float*)d_in[18], *b1 = (const float*)d_in[19];
  const float* w2 = (const float*)d_in[20], *b2 = (const float*)d_in[21];
  const float* fc1_w = (const float*)d_in[22], *fc1_b = (const float*)d_in[23];
  const float* fc2_w = (const float*)d_in[24], *fc2_b = (const float*)d_in[25];
  const float* fc3_w = (const float*)d_in[26], *fc3_b = (const float*)d_in[27];

  // workspace carve-up (256B aligned)
  char* p = (char*)d_ws;
  auto alloc = [&](size_t bytes) -> char* {
    char* r = p; p += (bytes + 255) & ~size_t(255); return r;
  };
  bstore* xb   = (bstore*)alloc((size_t)BSn * INn * 2);
  bstore* wT   = (bstore*)alloc((size_t)Dn * DFFn * 2);     // reused weight scratch
  float*  h_f  = (float*) alloc((size_t)BSn * Dn * 4);
  bstore* h_b  = (bstore*)alloc((size_t)BSn * Dn * 2);
  bstore* qB   = (bstore*)alloc((size_t)Sn * Dn * 2);
  bstore* kB   = (bstore*)alloc((size_t)Sn * Dn * 2);
  bstore* vTb  = (bstore*)alloc((size_t)Dn * Sn * 2);       // V stored transposed (D x S)
  float*  scF  = (float*) alloc((size_t)Hn * Sn * Sn * 4);
  bstore* awB  = (bstore*)alloc((size_t)Hn * Sn * Sn * 2);
  bstore* aoB  = (bstore*)alloc((size_t)Sn * Dn * 2);
  float*  aoF  = (float*) alloc((size_t)Sn * Dn * 4);
  float*  saF  = (float*) alloc((size_t)Sn * Dn * 4);
  bstore* ff1b = (bstore*)alloc((size_t)BSn * DFFn * 2);
  float*  ff2f = (float*) alloc((size_t)BSn * Dn * 4);
  bstore* t0   = (bstore*)alloc((size_t)BSn * HIDn * 2);
  bstore* t1   = (bstore*)alloc((size_t)BSn * HIDn * 2);
  int*    flag = (int*)   alloc(256);

  // input cast + mask flag
  long long nx = (long long)BSn * INn;
  k_cast_bf16<<<(unsigned)((nx + 255) / 256), 256, 0, stream>>>(x, xb, nx);
  k_mask<<<1, 256, 0, stream>>>(xm, Sn, flag);

  // h = x @ proj_w + proj_b + pos  (fp32 h + bf16 copy)
  launch_castT(stream, proj_w, wT, INn, Dn);
  launch_gemm(stream, xb, INn, 0, wT, INn, 0, proj_b, pos, (unsigned)(Sn - 1), Dn,
              h_f, Dn, 0, h_b, Dn, 0, nullptr, 0, 0,
              BSn, Dn, INn, 1.f, 0, 1);

  for (int i = 0; i < Ln; ++i) {
    const size_t wo = (size_t)i * Dn * Dn;
    // batch-0-only attention path (reference uses ao[0] only)
    launch_castT(stream, qw + wo, wT, Dn, Dn);
    launch_gemm(stream, h_b, Dn, 0, wT, Dn, 0, qb + i * Dn, nullptr, 0, 0,
                nullptr, 0, 0, qB, Dn, 0, nullptr, 0, 0,
                Sn, Dn, Dn, SCALEf, 0, 1);
    launch_castT(stream, kw + wo, wT, Dn, Dn);
    launch_gemm(stream, h_b, Dn, 0, wT, Dn, 0, kb + i * Dn, nullptr, 0, 0,
                nullptr, 0, 0, kB, Dn, 0, nullptr, 0, 0,
                Sn, Dn, Dn, 1.f, 0, 1);
    launch_castT(stream, vw + wo, wT, Dn, Dn);
    launch_gemm(stream, h_b, Dn, 0, wT, Dn, 0, vb + i * Dn, nullptr, 0, 0,
                nullptr, 0, 0, nullptr, 0, 0, vTb, Sn, 0,
                Sn, Dn, Dn, 1.f, 0, 1);
    // scores[h] = q_h @ k_h^T   (k rows serve directly as Bt rows)
    launch_gemm(stream, qB, Dn, HDn, kB, Dn, HDn, nullptr, nullptr, 0, 0,
                scF, Sn, (long long)Sn * Sn, nullptr, 0, 0, nullptr, 0, 0,
                Sn, Sn, HDn, 1.f, 0, Hn);
    // masked clamp-softmax -> bf16 attention weights
    k_softmax_row<<<Hn * Sn, 256, 0, stream>>>(scF, nullptr, awB, Sn, flag);
    // ao[h] = aw_h @ v_h   (vTb rows = v columns, contiguous)
    launch_gemm(stream, awB, Sn, (long long)Sn * Sn, vTb, Sn, (long long)HDn * Sn,
                nullptr, nullptr, 0, 0,
                nullptr, 0, 0, aoB, Dn, HDn, nullptr, 0, 0,
                Sn, HDn, Sn, 1.f, 0, Hn);
    // output projection
    launch_castT(stream, ow + wo, wT, Dn, Dn);
    launch_gemm(stream, aoB, Dn, 0, wT, Dn, 0, ob + i * Dn, nullptr, 0, 0,
                aoF, Dn, 0, nullptr, 0, 0, nullptr, 0, 0,
                Sn, Dn, Dn, 1.f, 0, 1);
    // clamp-softmax over embed dim -> sa (S,D), broadcast over batch in LN1
    k_softmax_row<<<Sn, 256, 0, stream>>>(aoF, saF, nullptr, Dn, nullptr);
    k_ln<<<BSn, 256, 0, stream>>>(h_f, saF, (unsigned)(Sn - 1),
                                  ln1_g + i * Dn, ln1_b + i * Dn, h_f, h_b, Dn);
    // FFN
    launch_castT(stream, w1 + (size_t)i * Dn * DFFn, wT, Dn, DFFn);
    launch_gemm(stream, h_b, Dn, 0, wT, Dn, 0, b1 + i * DFFn, nullptr, 0, 0,
                nullptr, 0, 0, ff1b, DFFn, 0, nullptr, 0, 0,
                BSn, DFFn, Dn, 1.f, 1, 1);
    launch_castT(stream, w2 + (size_t)i * DFFn * Dn, wT, DFFn, Dn);
    launch_gemm(stream, ff1b, DFFn, 0, wT, DFFn, 0, b2 + i * Dn, nullptr, 0, 0,
                ff2f, Dn, 0, nullptr, 0, 0, nullptr, 0, 0,
                BSn, Dn, DFFn, 1.f, 0, 1);
    k_ln<<<BSn, 256, 0, stream>>>(h_f, ff2f, 0xFFFFFFFFu,
                                  ln2_g + i * Dn, ln2_b + i * Dn, h_f, h_b, Dn);
  }

  // tail: fc1, fc2 x5, fc3
  launch_castT(stream, fc1_w, wT, Dn, HIDn);
  launch_gemm(stream, h_b, Dn, 0, wT, Dn, 0, fc1_b, nullptr, 0, 0,
              nullptr, 0, 0, t0, HIDn, 0, nullptr, 0, 0,
              BSn, HIDn, Dn, 1.f, 0, 1);
  launch_castT(stream, fc2_w, wT, HIDn, HIDn);
  bstore* ti = t0; bstore* to = t1;
  for (int r = 0; r < 5; ++r) {
    launch_gemm(stream, ti, HIDn, 0, wT, HIDn, 0, fc2_b, nullptr, 0, 0,
                nullptr, 0, 0, to, HIDn, 0, nullptr, 0, 0,
                BSn, HIDn, HIDn, 1.f, 0, 1);
    bstore* tmp = ti; ti = to; to = tmp;
  }
  launch_castT(stream, fc3_w, wT, HIDn, OUTn);
  launch_gemm(stream, ti, HIDn, 0, wT, HIDn, 0, fc3_b, nullptr, 0, 0,
              (float*)d_out, OUTn, 0, nullptr, 0, 0, nullptr, 0, 0,
              BSn, OUTn, HIDn, 1.f, 0, 1);
}
